// PointNetFeaturePropagation_70239895159141
// MI455X (gfx1250) — compile-verified, matrix operand order checked by reading, not run on hardware
//
#include <hip/hip_runtime.h>
#include <hip/hip_bf16.h>

// ---------------- problem constants ----------------
#define B_   8
#define N_   8192
#define S_   2048
#define D1_  128
#define D2_  256
#define CIN_ 384      // D1 + D2
#define C1_  256
#define C2_  128
#define NN_EPS 1e-8f
#define BN_EPS 1e-5f

typedef __attribute__((ext_vector_type(16))) _Float16 v16h;
typedef __attribute__((ext_vector_type(8)))  float    v8f;
typedef __attribute__((ext_vector_type(4)))  unsigned int v4u;

// ---------------------------------------------------------------------------
// weights fp32 -> fp16
// ---------------------------------------------------------------------------
__global__ void cvt_w_kernel(const float* __restrict__ w, _Float16* __restrict__ wh, int n) {
    int i = blockIdx.x * 256 + threadIdx.x;
    if (i < n) wh[i] = (_Float16)w[i];
}

__global__ void zero_kernel(float* __restrict__ p, int n) {
    int i = blockIdx.x * 256 + threadIdx.x;
    if (i < n) p[i] = 0.0f;
}

// ---------------------------------------------------------------------------
// 3-NN over S candidates per query point; squared distances, inv-dist weights.
// grid: (N/128, B), block 128. xyz layout [B,3,*].
// ---------------------------------------------------------------------------
__global__ __launch_bounds__(128) void knn_kernel(const float* __restrict__ xyz1,
                                                  const float* __restrict__ xyz2,
                                                  int* __restrict__ idx,
                                                  float* __restrict__ wgt) {
    __shared__ float sx[512], sy[512], sz[512];
    const int tid = threadIdx.x;
    const int b   = blockIdx.y;
    const int n   = blockIdx.x * 128 + tid;

    const float px = xyz1[(size_t)b * 3 * N_ + 0 * N_ + n];
    const float py = xyz1[(size_t)b * 3 * N_ + 1 * N_ + n];
    const float pz = xyz1[(size_t)b * 3 * N_ + 2 * N_ + n];

    float d0 = 3.4e38f, d1 = 3.4e38f, d2 = 3.4e38f;
    int   i0 = 0,       i1 = 0,       i2 = 0;

    const size_t xb = (size_t)b * 3 * S_;
    for (int s0 = 0; s0 < S_; s0 += 512) {
        for (int i = tid; i < 512; i += 128) {
            sx[i] = xyz2[xb + 0 * S_ + s0 + i];
            sy[i] = xyz2[xb + 1 * S_ + s0 + i];
            sz[i] = xyz2[xb + 2 * S_ + s0 + i];
        }
        __syncthreads();
        if (s0 + 512 < S_) {
            __builtin_prefetch(&xyz2[xb + s0 + 512], 0, 0);      // global_prefetch_b8
            __builtin_prefetch(&xyz2[xb + S_ + s0 + 512], 0, 0);
            __builtin_prefetch(&xyz2[xb + 2 * S_ + s0 + 512], 0, 0);
        }
#pragma unroll 4
        for (int j = 0; j < 512; ++j) {
            const float dx = px - sx[j];
            const float dy = py - sy[j];
            const float dz = pz - sz[j];
            const float d  = dx * dx + dy * dy + dz * dz;
            const int   s  = s0 + j;
            if (d < d0)      { d2 = d1; i2 = i1; d1 = d0; i1 = i0; d0 = d; i0 = s; }
            else if (d < d1) { d2 = d1; i2 = i1; d1 = d;  i1 = s; }
            else if (d < d2) { d2 = d;  i2 = s; }
        }
        __syncthreads();
    }

    const float r0 = 1.0f / (d0 + NN_EPS);
    const float r1 = 1.0f / (d1 + NN_EPS);
    const float r2 = 1.0f / (d2 + NN_EPS);
    const float rs = 1.0f / (r0 + r1 + r2);

    const size_t o = ((size_t)b * N_ + n) * 3;
    idx[o + 0] = i0; idx[o + 1] = i1; idx[o + 2] = i2;
    wgt[o + 0] = r0 * rs; wgt[o + 1] = r1 * rs; wgt[o + 2] = r2 * rs;
}

// ---------------------------------------------------------------------------
// Build concatenated activation tensor Xcat (f16) [B, CIN, N]:
//   k <  D1 : copy of points1
//   k >= D1 : weighted 3-NN gather from points2 channel row staged in LDS
// grid: (N/1024, CIN, B), block 256 (4 points per thread -> row staged 8x
// total instead of 32x).
// ---------------------------------------------------------------------------
__global__ __launch_bounds__(256) void gather_concat_kernel(const float* __restrict__ points1,
                                                            const float* __restrict__ points2,
                                                            const int* __restrict__ idx,
                                                            const float* __restrict__ wgt,
                                                            _Float16* __restrict__ xcat) {
    __shared__ float row[S_];
    const int tid = threadIdx.x;
    const int k   = blockIdx.y;
    const int b   = blockIdx.z;
    const int n0  = blockIdx.x * 1024;

    if (k < D1_) {
        const size_t src = ((size_t)b * D1_ + k) * N_;
        const size_t dst = ((size_t)b * CIN_ + k) * N_;
#pragma unroll
        for (int p = 0; p < 4; ++p) {
            const int n = n0 + p * 256 + tid;
            xcat[dst + n] = (_Float16)points1[src + n];
        }
        return;
    }
    const int d = k - D1_;
    for (int i = tid; i < S_; i += 256)
        row[i] = points2[((size_t)b * D2_ + d) * S_ + i];
    __syncthreads();

    const size_t dst = ((size_t)b * CIN_ + k) * N_;
#pragma unroll
    for (int p = 0; p < 4; ++p) {
        const int n = n0 + p * 256 + tid;
        const size_t o = ((size_t)b * N_ + n) * 3;
        const float v = wgt[o + 0] * row[idx[o + 0]] +
                        wgt[o + 1] * row[idx[o + 1]] +
                        wgt[o + 2] * row[idx[o + 2]];
        xcat[dst + n] = (_Float16)v;
    }
}

// ---------------------------------------------------------------------------
// WMMA GEMM:  Yh[b][m][n] = f16( sum_k Wh[m][k] * Xh[b][k][n] + bias[m] )
// Block = 256 threads = 8 waves. Block tile: M=128 x N=64.
// Wave w owns the 16-row strip m = w*16..w*16+15 and all 4 N-subtiles:
// one A fragment reused across 4 v_wmma_f32_16x16x32_f16 per K-step.
// grid: (Nn/64, Mtot/128, B)
// ---------------------------------------------------------------------------
#define ASTR 40   // 32 + 8 halfs padding
#define BSTR 40

__global__ __launch_bounds__(256) void gemm_wmma_kernel(const _Float16* __restrict__ Xh,
                                                        const _Float16* __restrict__ Wh,
                                                        const float* __restrict__ bias,
                                                        _Float16* __restrict__ Yh,
                                                        int Mtot, int Ktot, int Nn) {
    __shared__ _Float16 As[128 * ASTR];   // A tile 128 x 32 (row-major, padded)
    __shared__ _Float16 Bs[64 * BSTR];    // B tile transposed: [n][k] 64 x 32

    const int tid   = threadIdx.x;
    const int wid   = tid >> 5;
    const int lane  = tid & 31;
    const int mm    = lane & 15;         // M (A) / N (B) within 16x16 tile
    const int hi    = lane >> 4;         // selects K halves per ISA layout
    const int kb    = hi * 8;

    const int nBase = blockIdx.x * 64;
    const int mBase = blockIdx.y * 128;
    const int b     = blockIdx.z;

    v8f acc0 = {}, acc1 = {}, acc2 = {}, acc3 = {};

    for (int kt = 0; kt < Ktot; kt += 32) {
        // ---- A tile: 128 rows x 32 halfs; 512 x 16B vectors, 2 per thread
#pragma unroll
        for (int i = 0; i < 2; ++i) {
            const int lin = tid + i * 256;         // 0..511
            const int r   = lin >> 2;              // 0..127
            const int c4  = lin & 3;               // 0..3 (x8 halfs)
            const v4u v = *reinterpret_cast<const v4u*>(
                &Wh[(size_t)(mBase + r) * Ktot + kt + c4 * 8]);
            *reinterpret_cast<v4u*>(&As[r * ASTR + c4 * 8]) = v;
        }
        // ---- B tile: 32 k-rows x 64 halfs, coalesced dword reads,
        //      transposed into Bs[n][k]; 1024 dwords -> 4 per thread
#pragma unroll
        for (int i = 0; i < 4; ++i) {
            const int lin = tid + i * 256;         // 0..1023
            const int r   = lin >> 5;              // k row 0..31
            const int dw  = lin & 31;              // dword in row (two n's)
            union { unsigned int u; _Float16 h[2]; } cv;
            cv.u = *reinterpret_cast<const unsigned int*>(
                &Xh[((size_t)b * Ktot + kt + r) * Nn + nBase + dw * 2]);
            Bs[(dw * 2 + 0) * BSTR + r] = cv.h[0];
            Bs[(dw * 2 + 1) * BSTR + r] = cv.h[1];
        }
        __syncthreads();

        // ---- fragments per documented 16-bit A/B layouts:
        // lanes 0-15: K = {0..7, 16..23}; lanes 16-31: K = {8..15, 24..31}
        union AV { v4u u[2]; v16h h; } av, bv0, bv1, bv2, bv3;
        av.u[0] = *reinterpret_cast<const v4u*>(&As[(wid * 16 + mm) * ASTR + kb]);
        av.u[1] = *reinterpret_cast<const v4u*>(&As[(wid * 16 + mm) * ASTR + kb + 16]);
        bv0.u[0] = *reinterpret_cast<const v4u*>(&Bs[( 0 + mm) * BSTR + kb]);
        bv0.u[1] = *reinterpret_cast<const v4u*>(&Bs[( 0 + mm) * BSTR + kb + 16]);
        bv1.u[0] = *reinterpret_cast<const v4u*>(&Bs[(16 + mm) * BSTR + kb]);
        bv1.u[1] = *reinterpret_cast<const v4u*>(&Bs[(16 + mm) * BSTR + kb + 16]);
        bv2.u[0] = *reinterpret_cast<const v4u*>(&Bs[(32 + mm) * BSTR + kb]);
        bv2.u[1] = *reinterpret_cast<const v4u*>(&Bs[(32 + mm) * BSTR + kb + 16]);
        bv3.u[0] = *reinterpret_cast<const v4u*>(&Bs[(48 + mm) * BSTR + kb]);
        bv3.u[1] = *reinterpret_cast<const v4u*>(&Bs[(48 + mm) * BSTR + kb + 16]);

        acc0 = __builtin_amdgcn_wmma_f32_16x16x32_f16(false, av.h, false, bv0.h,
                                                      (short)0, acc0, false, false);
        acc1 = __builtin_amdgcn_wmma_f32_16x16x32_f16(false, av.h, false, bv1.h,
                                                      (short)0, acc1, false, false);
        acc2 = __builtin_amdgcn_wmma_f32_16x16x32_f16(false, av.h, false, bv2.h,
                                                      (short)0, acc2, false, false);
        acc3 = __builtin_amdgcn_wmma_f32_16x16x32_f16(false, av.h, false, bv3.h,
                                                      (short)0, acc3, false, false);

        __syncthreads();
    }

    // D layout: lane l, vgpr r -> M = r + (l>=16 ? 8 : 0), N = l & 15
    const int chBase = mBase + wid * 16 + hi * 8;
    const float bias0 = bias[chBase + 0], bias1 = bias[chBase + 1];
    const float bias2 = bias[chBase + 2], bias3 = bias[chBase + 3];
    const float bias4 = bias[chBase + 4], bias5 = bias[chBase + 5];
    const float bias6 = bias[chBase + 6], bias7 = bias[chBase + 7];
    const float biasv[8] = {bias0, bias1, bias2, bias3, bias4, bias5, bias6, bias7};

    v8f* accs[4] = {&acc0, &acc1, &acc2, &acc3};
#pragma unroll
    for (int j = 0; j < 4; ++j) {
        const int n = nBase + j * 16 + mm;
#pragma unroll
        for (int r = 0; r < 8; ++r) {
            const int ch = chBase + r;
            const float v = (*accs[j])[r] + biasv[r];
            Yh[((size_t)b * Mtot + ch) * Nn + n] = (_Float16)v;
        }
    }
}

// ---------------------------------------------------------------------------
// Per-channel sum / sumsq over (B, N): block = (channel, batch), LDS tree
// reduce, one atomicAdd pair per block. stats[ch] = sum, stats[M+ch] = sumsq.
// ---------------------------------------------------------------------------
__global__ __launch_bounds__(256) void stats_kernel(const _Float16* __restrict__ Y,
                                                    float* __restrict__ stats,
                                                    int Mtot, int Nn) {
    __shared__ float rs[256], rq[256];
    const int tid = threadIdx.x;
    const int ch  = blockIdx.x;
    const int b   = blockIdx.y;
    const size_t base = ((size_t)b * Mtot + ch) * Nn;

    float s = 0.0f, q = 0.0f;
    for (int i = tid; i < Nn; i += 256) {
        const float v = (float)Y[base + i];
        s += v; q += v * v;
    }
    rs[tid] = s; rq[tid] = q;
    __syncthreads();
    for (int st = 128; st > 0; st >>= 1) {
        if (tid < st) { rs[tid] += rs[tid + st]; rq[tid] += rq[tid + st]; }
        __syncthreads();
    }
    if (tid == 0) {
        atomicAdd(&stats[ch], rs[0]);
        atomicAdd(&stats[Mtot + ch], rq[0]);
    }
}

__global__ void finalize_stats_kernel(const float* __restrict__ stats,
                                      float* __restrict__ mr, int Mtot, float invCount) {
    const int ch = blockIdx.x * 256 + threadIdx.x;
    if (ch >= Mtot) return;
    const float mean = stats[ch] * invCount;
    const float var  = stats[Mtot + ch] * invCount - mean * mean;
    mr[ch]        = mean;
    mr[Mtot + ch] = rsqrtf(var + BN_EPS);
}

// ---------------------------------------------------------------------------
// BN (training-mode, batch stats) + ReLU. Writes f16 (inter-layer) or f32
// (final output) depending on which pointer is non-null.
// grid: (Nn/256, Mtot, B)
// ---------------------------------------------------------------------------
__global__ __launch_bounds__(256) void bn_relu_kernel(const _Float16* __restrict__ Y,
                                                      const float* __restrict__ mr,
                                                      const float* __restrict__ gamma,
                                                      const float* __restrict__ beta,
                                                      _Float16* __restrict__ outh,
                                                      float* __restrict__ outf,
                                                      int Mtot, int Nn) {
    const int ch = blockIdx.y;
    const int b  = blockIdx.z;
    const int n  = blockIdx.x * 256 + threadIdx.x;
    const size_t o = ((size_t)b * Mtot + ch) * Nn + n;

    float v = (float)Y[o];
    v = (v - mr[ch]) * mr[Mtot + ch] * gamma[ch] + beta[ch];
    v = fmaxf(v, 0.0f);
    if (outh) outh[o] = (_Float16)v;
    else      outf[o] = v;
}

// ---------------------------------------------------------------------------
// launch
// ---------------------------------------------------------------------------
extern "C" void kernel_launch(void* const* d_in, const int* in_sizes, int n_in,
                              void* d_out, int out_size, void* d_ws, size_t ws_size,
                              hipStream_t stream) {
    const float* xyz1    = (const float*)d_in[0];
    const float* xyz2    = (const float*)d_in[1];
    const float* points1 = (const float*)d_in[2];
    const float* points2 = (const float*)d_in[3];
    const float* w1      = (const float*)d_in[4];
    const float* b1      = (const float*)d_in[5];
    const float* gamma1  = (const float*)d_in[6];
    const float* beta1   = (const float*)d_in[7];
    const float* w2      = (const float*)d_in[8];
    const float* b2      = (const float*)d_in[9];
    const float* gamma2  = (const float*)d_in[10];
    const float* beta2   = (const float*)d_in[11];
    float* out = (float*)d_out;

    // ---- carve workspace
    char* ws = (char*)d_ws;
    size_t off = 0;
    auto take = [&](size_t bytes) -> char* {
        char* p = ws + off;
        off += (bytes + 255) & ~(size_t)255;
        return p;
    };
    int*       idx  = (int*)      take((size_t)B_ * N_ * 3 * sizeof(int));
    float*     wgt  = (float*)    take((size_t)B_ * N_ * 3 * sizeof(float));
    _Float16*  w1h  = (_Float16*) take((size_t)C1_ * CIN_ * 2);
    _Float16*  w2h  = (_Float16*) take((size_t)C2_ * C1_ * 2);
    _Float16*  xcat = (_Float16*) take((size_t)B_ * CIN_ * N_ * 2);
    _Float16*  y1   = (_Float16*) take((size_t)B_ * C1_ * N_ * 2);
    _Float16*  x2   = (_Float16*) take((size_t)B_ * C1_ * N_ * 2);
    _Float16*  y2   = (_Float16*) take((size_t)B_ * C2_ * N_ * 2);
    float*     st1  = (float*)    take(2 * C1_ * sizeof(float));
    float*     mr1  = (float*)    take(2 * C1_ * sizeof(float));
    float*     st2  = (float*)    take(2 * C2_ * sizeof(float));
    float*     mr2  = (float*)    take(2 * C2_ * sizeof(float));

    // ---- weights to f16
    cvt_w_kernel<<<dim3((C1_ * CIN_ + 255) / 256), 256, 0, stream>>>(w1, w1h, C1_ * CIN_);
    cvt_w_kernel<<<dim3((C2_ * C1_ + 255) / 256), 256, 0, stream>>>(w2, w2h, C2_ * C1_);

    // ---- 3-NN + weights
    knn_kernel<<<dim3(N_ / 128, B_), 128, 0, stream>>>(xyz1, xyz2, idx, wgt);

    // ---- gather + concat -> Xcat f16 [B, 384, N]
    gather_concat_kernel<<<dim3(N_ / 1024, CIN_, B_), 256, 0, stream>>>(
        points1, points2, idx, wgt, xcat);

    // ---- layer 1: GEMM (WMMA) + BN + ReLU
    zero_kernel<<<dim3((2 * C1_ + 255) / 256), 256, 0, stream>>>(st1, 2 * C1_);
    gemm_wmma_kernel<<<dim3(N_ / 64, C1_ / 128, B_), 256, 0, stream>>>(
        xcat, w1h, b1, y1, C1_, CIN_, N_);
    stats_kernel<<<dim3(C1_, B_), 256, 0, stream>>>(y1, st1, C1_, N_);
    finalize_stats_kernel<<<dim3((C1_ + 255) / 256), 256, 0, stream>>>(
        st1, mr1, C1_, 1.0f / (float)((size_t)B_ * N_));
    bn_relu_kernel<<<dim3(N_ / 256, C1_, B_), 256, 0, stream>>>(
        y1, mr1, gamma1, beta1, x2, nullptr, C1_, N_);

    // ---- layer 2: GEMM (WMMA) + BN + ReLU -> d_out (f32)
    zero_kernel<<<dim3((2 * C2_ + 255) / 256), 256, 0, stream>>>(st2, 2 * C2_);
    gemm_wmma_kernel<<<dim3(N_ / 64, C2_ / 128, B_), 256, 0, stream>>>(
        x2, w2h, b2, y2, C2_, C1_, N_);
    stats_kernel<<<dim3(C2_, B_), 256, 0, stream>>>(y2, st2, C2_, N_);
    finalize_stats_kernel<<<dim3((C2_ + 255) / 256), 256, 0, stream>>>(
        st2, mr2, C2_, 1.0f / (float)((size_t)B_ * N_));
    bn_relu_kernel<<<dim3(N_ / 256, C2_, B_), 256, 0, stream>>>(
        y2, mr2, gamma2, beta2, nullptr, out, C2_, N_);
}
